// FullyConnectedTEDecoder_27135603376524
// MI455X (gfx1250) — matrix operand order repaired and checked by reading, not compiled, not run on hardware
//
#include <hip/hip_runtime.h>
#include <hip/hip_bf16.h>

typedef __attribute__((ext_vector_type(16))) _Float16 v16h;
typedef __attribute__((ext_vector_type(8)))  float    v8f;
typedef __attribute__((ext_vector_type(4)))  float    v4f;
typedef __attribute__((ext_vector_type(4)))  unsigned int v4u;

#define TB 4
#define TN 384
#define TH 128
#define TDOUT 64
#define MTILES (TN / 16)          // 24 m-tiles per (b,n)
#define NWAVES (TB * TN * MTILES) // 36864 total wave-tiles

// ---------------------------------------------------------------------------
// P1: c[b][j] = b1[j] + sum_d r[b][d] * W1[2+d][j]      (4 x 128)
// ---------------------------------------------------------------------------
__global__ __launch_bounds__(512) void prep_c_kernel(
    const float* __restrict__ r, const float* __restrict__ W1,
    const float* __restrict__ b1, float* __restrict__ c) {
  int t = threadIdx.x;          // 512 threads: b = t/128, j = t%128
  int b = t >> 7;
  int j = t & 127;
  float acc = b1[j];
  const float* rb = r + b * 128;
#pragma unroll 8
  for (int d = 0; d < 128; ++d)
    acc += rb[d] * W1[(2 + d) * 128 + j];
  c[t] = acc;
}

// ---------------------------------------------------------------------------
// P2: a[b][n][j] = x_ctx[b][n][0]*W1[0][j] + x_ctx[b][n][1]*W1[1][j]
// ---------------------------------------------------------------------------
__global__ __launch_bounds__(512) void prep_a_kernel(
    const float* __restrict__ x_ctx, const float* __restrict__ W1,
    float* __restrict__ a) {
  int idx = blockIdx.x * 512 + threadIdx.x;   // 196608 threads
  int j = idx & 127;
  int bn = idx >> 7;                          // b*384 + n
  float x0 = x_ctx[bn * 2 + 0];
  float x1 = x_ctx[bn * 2 + 1];
  a[idx] = x0 * W1[j] + x1 * W1[128 + j];
}

// ---------------------------------------------------------------------------
// P3: pack W2 (f32 [128][64]) into f16 B-operand layout.
// index t = ((kc*4 + nt)*32 + lane)*8 + v ; each uint32 = half2{W2[k][col], W2[k+1][col]}
// with k = kc*32 + (lane/16)*8 + 2v + (v/4)*8, col = nt*16 + lane%16.
// ---------------------------------------------------------------------------
__global__ __launch_bounds__(512) void prep_w2_kernel(
    const float* __restrict__ W2, unsigned int* __restrict__ w2p) {
  int t = blockIdx.x * 512 + threadIdx.x;     // 4096 threads
  int v  = t & 7;
  int L  = (t >> 3) & 31;
  int nt = (t >> 8) & 3;
  int kc = t >> 10;
  int k   = kc * 32 + ((L >> 4) << 3) + (v << 1) + ((v >> 2) << 3);
  int col = (nt << 4) + (L & 15);
  union { _Float16 h[2]; unsigned int u; } pk;
  pk.h[0] = (_Float16)W2[k * TDOUT + col];
  pk.h[1] = (_Float16)W2[(k + 1) * TDOUT + col];
  w2p[t] = pk.u;
}

// ---------------------------------------------------------------------------
// Main: per wave, one 16-pair x 64-output tile.
//   h[row][k] = relu(a[b,n][k] - a[b,m0+row][k] + c[b][k])  (built in A layout)
//   out tile  = h @ W2 + b2 via v_wmma_f32_16x16x32_f16 (K = 4 chunks of 32)
// ---------------------------------------------------------------------------
__global__ __launch_bounds__(128) void te_decoder_main_kernel(
    const float* __restrict__ aTab,        // [B*N][128]
    const float* __restrict__ cTab,        // [B][128]
    const unsigned int* __restrict__ w2p,  // packed B operands (16 KB)
    const float* __restrict__ b2,          // [64]
    float* __restrict__ out) {             // [B*N*N][64]
  const int wave = (blockIdx.x << 2) + (threadIdx.x >> 5);
  const int lane = threadIdx.x & 31;

  const int mt = wave % MTILES;
  const int n  = (wave / MTILES) % TN;
  const int b  = wave / (MTILES * TN);
  const int m0 = mt << 4;

  const int row = lane & 15;
  const int hi  = lane >> 4;
  const int m   = m0 + row;

  const float* an = aTab + (b * TN + n) * TH;
  const float* am = aTab + (b * TN + m) * TH;
  const float* cb = cTab + b * TH;

  // Build all four A-operand chunks (f16, ISA 16-bit A 16x32 layout).
  v16h A[4];
#pragma unroll
  for (int kc = 0; kc < 4; ++kc) {
    const int k0 = kc * 32 + (hi << 3);  // V0..V3 cover k0..k0+7
    const int k1 = k0 + 16;              // V4..V7 cover k1..k1+7
    v4f an0 = *(const v4f*)(an + k0);
    v4f an1 = *(const v4f*)(an + k0 + 4);
    v4f an2 = *(const v4f*)(an + k1);
    v4f an3 = *(const v4f*)(an + k1 + 4);
    v4f am0 = *(const v4f*)(am + k0);
    v4f am1 = *(const v4f*)(am + k0 + 4);
    v4f am2 = *(const v4f*)(am + k1);
    v4f am3 = *(const v4f*)(am + k1 + 4);
    v4f c0  = *(const v4f*)(cb + k0);
    v4f c1  = *(const v4f*)(cb + k0 + 4);
    v4f c2  = *(const v4f*)(cb + k1);
    v4f c3  = *(const v4f*)(cb + k1 + 4);
#pragma unroll
    for (int i = 0; i < 4; ++i) {
      A[kc][i]      = (_Float16)fmaxf(an0[i] - am0[i] + c0[i], 0.0f);
      A[kc][4 + i]  = (_Float16)fmaxf(an1[i] - am1[i] + c1[i], 0.0f);
      A[kc][8 + i]  = (_Float16)fmaxf(an2[i] - am2[i] + c2[i], 0.0f);
      A[kc][12 + i] = (_Float16)fmaxf(an3[i] - am3[i] + c3[i], 0.0f);
    }
  }

  const long rowBase = (long)((b * TN + n) * TN + m0) * TDOUT;

#pragma unroll
  for (int nt = 0; nt < 4; ++nt) {
    v8f acc = {};
#pragma unroll
    for (int kc = 0; kc < 4; ++kc) {
      const unsigned int* bp = w2p + ((((kc << 2) + nt) << 5) + lane) * 8;
      union { v4u u[2]; v16h h; } bu;
      bu.u[0] = *(const v4u*)(bp);
      bu.u[1] = *(const v4u*)(bp + 4);
      acc = __builtin_amdgcn_wmma_f32_16x16x32_f16(
          false, A[kc], false, bu.h, (short)0, acc, false, false);
    }
    const int col = (nt << 4) + row;
    const float bias = b2[col];
    float* op = out + rowBase + col;
#pragma unroll
    for (int j = 0; j < 8; ++j) {
      const int r2 = (hi << 3) + j;   // C/D layout: lanes>=16 hold M = j+8
      __builtin_nontemporal_store(acc[j] + bias, op + (long)r2 * TDOUT);
    }
  }
}

// ---------------------------------------------------------------------------
extern "C" void kernel_launch(void* const* d_in, const int* in_sizes, int n_in,
                              void* d_out, int out_size, void* d_ws, size_t ws_size,
                              hipStream_t stream) {
  const float* r     = (const float*)d_in[0];  // [4,128]
  const float* x_ctx = (const float*)d_in[1];  // [4,384,2]
  // d_in[2] = y_ctx (unused), d_in[3] = x_trg (unused)
  const float* W1 = (const float*)d_in[4];     // [130,128]
  const float* b1 = (const float*)d_in[5];     // [128]
  const float* W2 = (const float*)d_in[6];     // [128,64]
  const float* b2 = (const float*)d_in[7];     // [64]
  float* out = (float*)d_out;                  // [4,384,384,64]

  char* ws = (char*)d_ws;
  float* aTab = (float*)ws;                                  // 786432 B
  float* cTab = (float*)(ws + 786432);                       //   2048 B
  unsigned int* w2p = (unsigned int*)(ws + 786432 + 2048);   //  16384 B

  prep_c_kernel<<<1, 512, 0, stream>>>(r, W1, b1, cTab);
  prep_a_kernel<<<TB * TN * TH / 512, 512, 0, stream>>>(x_ctx, W1, aTab);
  prep_w2_kernel<<<8, 512, 0, stream>>>(W2, w2p);

  te_decoder_main_kernel<<<NWAVES / 4, 128, 0, stream>>>(aTab, cTab, w2p, b2, out);
}
